// Unimodal_DecoderLSTM_61065845014978
// MI455X (gfx1250) — compile-verified, matrix operand order checked by reading, not run on hardware
//
#include <hip/hip_runtime.h>
#include <hip/hip_bf16.h>

typedef _Float16 h8   __attribute__((ext_vector_type(8)));
typedef _Float16 v16h __attribute__((ext_vector_type(16)));
typedef float    v8f  __attribute__((ext_vector_type(8)));

#define PRED_LEN 30
#define H_DIM    128
#define EMB      16
#define BATCH    65536
#define ROWS_PER_WAVE  32
#define ROWS_PER_BLOCK 256   // 8 waves * 32 rows

// LDS layout (halfs first for alignment, then floats)
#define WHH_HALFS   (32*4*32*16)        // 32 n-tiles * 4 k-chunks * 32 lanes * 16 halfs = 65536
#define WIH_HALFS   (32*32*16)          // 32 n-tiles * 32 lanes * 16 halfs (K16 = bias) = 16384
#define WOUT_HALFS  (4*32*16)           // 4 k-chunks * 32 lanes * 16 halfs (N padded)   = 2048
#define HSTG_HALFS  (8*ROWS_PER_WAVE*H_DIM)  // 8 waves * 32 rows * 128 cols             = 32768
#define SMEM_FLOATS (8*64 + 32 + 16 + 2)
#define SMEM_BYTES  ((WHH_HALFS + WIH_HALFS + WOUT_HALFS + HSTG_HALFS)*2 + SMEM_FLOATS*4 + 64)

__device__ __forceinline__ float hw_tanh(float x) {
#if __has_builtin(__builtin_amdgcn_tanhf)
    return __builtin_amdgcn_tanhf(x);           // v_tanh_f32 (CDNA5 TRANS op)
#elif __has_builtin(__builtin_amdgcn_tanh_f32)
    return __builtin_amdgcn_tanh_f32(x);
#else
    float e = __expf(2.0f * x);
    return 1.0f - 2.0f / (e + 1.0f);
#endif
}
__device__ __forceinline__ float hw_sigmoid(float x) {
    return 0.5f * hw_tanh(0.5f * x) + 0.5f;
}

__device__ __forceinline__ v8f wmma16(v16h a, v16h b, v8f c) {
    // D = A(16x32 f16) * B(32x16 f16) + C(16x16 f32)
    return __builtin_amdgcn_wmma_f32_16x16x32_f16(
        /*neg_a=*/false, a, /*neg_b=*/false, b,
        /*c_mod=*/(short)0, c, /*reuse_a=*/false, /*reuse_b=*/false);
}

__device__ __forceinline__ v16h assemble16(const _Float16* p0, const _Float16* p1) {
    h8 a = *(const h8*)p0;
    h8 b = *(const h8*)p1;
    v16h r;
#pragma unroll
    for (int i = 0; i < 8; ++i) { r[i] = a[i]; r[i + 8] = b[i]; }
    return r;
}

__global__ __launch_bounds__(256)
void lstm_decoder_kernel(const float* __restrict__ rel_in,   // last_obs_rel (B,2)
                         const float* __restrict__ h0,       // (B,128)
                         const float* __restrict__ c0,       // (B,128)
                         const float* __restrict__ Wsp,      // (16,2)
                         const float* __restrict__ bsp,      // (16)
                         const float* __restrict__ Wih,      // (512,16)
                         const float* __restrict__ bih,      // (512)
                         const float* __restrict__ Whh,      // (512,128)
                         const float* __restrict__ bhh,      // (512)
                         const float* __restrict__ Wout,     // (2,128)
                         const float* __restrict__ bout_,    // (2)
                         float* __restrict__ out)            // (B,30,2)
{
    extern __shared__ char smem[];
    _Float16* whhP  = (_Float16*)smem;               // packed B-fragments of W_hh^T (f16)
    _Float16* wihP  = whhP + WHH_HALFS;              // packed B-fragments of W_ih^T (+bias @K16)
    _Float16* woutP = wihP + WIH_HALFS;              // packed B-fragments of W_out^T (N-padded)
    _Float16* hstg  = woutP + WOUT_HALFS;            // per-wave h staging, row-major [32][128]
    float* relS  = (float*)(hstg + HSTG_HALFS);      // 8*64 : per-wave rel_pos [d*32+row]
    float* wspS  = relS + 8*64;                      // 32
    float* bspS  = wspS + 32;                        // 16
    float* boutS = bspS + 16;                        // 2

    const int tid = threadIdx.x;

    // ---------- one-time: pack weights into per-lane fragment order ----------
    // B 32x16 f16 layout assumed (mirror of documented A 16x32):
    //   lanes 0-15 : vgpr0-3 = K 0..7,  vgpr4-7 = K 16..23  (N = lane%16)
    //   lanes16-31 : vgpr0-3 = K 8..15, vgpr4-7 = K 24..31
    for (int idx = tid; idx < WHH_HALFS; idx += 256) {
        int e    = idx & 15;
        int lane = (idx >> 4) & 31;
        int kc   = (idx >> 9) & 3;
        int tile = idx >> 11;
        int klocal = (e < 8 ? e : e + 8) + ((lane >= 16) ? 8 : 0);
        int k = kc * 32 + klocal;
        int n = tile * 16 + (lane & 15);
        whhP[idx] = (_Float16)Whh[n * H_DIM + k];
    }
    // W_ih fragments: K0..15 = W_ih, K16 = (b_ih + b_hh)  (paired with x[K16] = 1.0)
    for (int idx = tid; idx < WIH_HALFS; idx += 256) {
        int e    = idx & 15;
        int lane = (idx >> 4) & 31;
        int tile = idx >> 9;
        int klocal = (e < 8 ? e : e + 8) + ((lane >= 16) ? 8 : 0);
        int n = tile * 16 + (lane & 15);
        float v = 0.0f;
        if (klocal < EMB)        v = Wih[n * EMB + klocal];
        else if (klocal == EMB)  v = bih[n] + bhh[n];
        wihP[idx] = (_Float16)v;
    }
    // W_out fragments: N 0..1 real, 2..15 zero
    for (int idx = tid; idx < WOUT_HALFS; idx += 256) {
        int e    = idx & 15;
        int lane = (idx >> 4) & 31;
        int kc   = idx >> 9;
        int klocal = (e < 8 ? e : e + 8) + ((lane >= 16) ? 8 : 0);
        int k = kc * 32 + klocal;
        int n = lane & 15;
        woutP[idx] = (n < 2) ? (_Float16)Wout[n * H_DIM + k] : (_Float16)0.0f;
    }
    if (tid < 32)  wspS[tid]  = Wsp[tid];
    if (tid < 16)  bspS[tid]  = bsp[tid];
    if (tid < 2)   boutS[tid] = bout_[tid];
    __syncthreads();

    // ---------- per-wave state ----------
    const int wave = tid >> 5;
    const int lane = tid & 31;
    const int m  = lane & 15;    // N within tile / row index for projections
    const int hi = lane >> 4;    // half-wave select
    const int rowbase = blockIdx.x * ROWS_PER_BLOCK + wave * ROWS_PER_WAVE;
    _Float16* hw   = hstg + wave * (ROWS_PER_WAVE * H_DIM);
    float*    relw = relS + wave * 64;
    const v8f vzero = {0.f,0.f,0.f,0.f,0.f,0.f,0.f,0.f};

    // h0 -> f16 staging (row-major, 32 rows)
    for (int idx = lane; idx < ROWS_PER_WAVE * H_DIM; idx += 32)
        hw[idx] = (_Float16)h0[(size_t)rowbase * H_DIM + idx];

    // c0 -> register D-tiles: cacc[s][j][r] = (row = s*16 + r + 8*hi, col = j*16 + m)
    v8f cacc[2][8];
#pragma unroll
    for (int s = 0; s < 2; ++s)
#pragma unroll
        for (int j = 0; j < 8; ++j)
#pragma unroll
            for (int r = 0; r < 8; ++r)
                cacc[s][j][r] = c0[(size_t)(rowbase + s*16 + r + 8*hi) * H_DIM + j*16 + m];

    // initial x = tanh(last_obs_rel @ Wsp^T + bsp) in A-fragment layout (per row-tile).
    // K16 slot (lane<16, elem 8) carries 1.0 to multiply the bias row at K16 of W_ih.
    v16h xfrag[2];
#pragma unroll
    for (int s = 0; s < 2; ++s) {
        float r0 = rel_in[(size_t)(rowbase + s*16 + m) * 2 + 0];
        float r1 = rel_in[(size_t)(rowbase + s*16 + m) * 2 + 1];
#pragma unroll
        for (int ee = 0; ee < 8; ++ee) {
            int e = hi * 8 + ee;
            float v = hw_tanh(r0 * wspS[e*2+0] + r1 * wspS[e*2+1] + bspS[e]);
            xfrag[s][ee]     = (_Float16)v;
            xfrag[s][ee + 8] = (_Float16)0.0f;
        }
        xfrag[s][8] = (hi == 0) ? (_Float16)1.0f : (_Float16)0.0f;   // bias lane (K=16)
    }

    // preload h A-fragments (2 row-tiles x 4 k-chunks)
    v16h afrag[2][4];
#pragma unroll
    for (int s = 0; s < 2; ++s)
#pragma unroll
        for (int kc = 0; kc < 4; ++kc) {
            const _Float16* p0 = hw + (s*16 + m) * H_DIM + kc * 32 + hi * 8;
            afrag[s][kc] = assemble16(p0, p0 + 16);
        }

    // ---------- recurrence ----------
#pragma unroll 1
    for (int t = 0; t < PRED_LEN; ++t) {
        // gates = x @ Wih^T (+bias via K16) + h @ Whh^T ; gate order i,f,g,o
        // Each B fragment is loaded once and feeds BOTH 16-row A tiles.
#pragma unroll
        for (int j = 0; j < 8; ++j) {
            v8f acc[4][2];
#pragma unroll
            for (int g = 0; g < 4; ++g) {
                const int tile = g * 8 + j;              // N-tile within 512-wide gates
                const _Float16* ip = wihP + ((size_t)tile * 32 + lane) * 16;
                v16h bI = assemble16(ip, ip + 8);
                acc[g][0] = wmma16(xfrag[0], bI, vzero);
                acc[g][1] = wmma16(xfrag[1], bI, vzero);
#pragma unroll
                for (int kc = 0; kc < 4; ++kc) {
                    const _Float16* bp = whhP + ((size_t)(tile * 4 + kc) * 32 + lane) * 16;
                    v16h bW = assemble16(bp, bp + 8);
                    acc[g][0] = wmma16(afrag[0][kc], bW, acc[g][0]);
                    acc[g][1] = wmma16(afrag[1][kc], bW, acc[g][1]);
                }
            }
            // elementwise LSTM cell update for hidden cols j*16 + m, both row-tiles
#pragma unroll
            for (int s = 0; s < 2; ++s)
#pragma unroll
                for (int r = 0; r < 8; ++r) {
                    float iv = hw_sigmoid(acc[0][s][r]);
                    float fv = hw_sigmoid(acc[1][s][r]);
                    float gv = hw_tanh(acc[2][s][r]);
                    float ov = hw_sigmoid(acc[3][s][r]);
                    float cn = fv * cacc[s][j][r] + iv * gv;
                    cacc[s][j][r] = cn;
                    float hn = ov * hw_tanh(cn);
                    hw[(s*16 + r + 8*hi) * H_DIM + j*16 + m] = (_Float16)hn;
                }
        }

        // reload h A-fragments with the new h (used by rel-WMMA and next step)
#pragma unroll
        for (int s = 0; s < 2; ++s)
#pragma unroll
            for (int kc = 0; kc < 4; ++kc) {
                const _Float16* p0 = hw + (s*16 + m) * H_DIM + kc * 32 + hi * 8;
                afrag[s][kc] = assemble16(p0, p0 + 16);
            }

        // rel_pos = h_new @ Wout^T + b_out  (WMMA against N-padded W_out)
        v8f racc0 = vzero, racc1 = vzero;
#pragma unroll
        for (int kc = 0; kc < 4; ++kc) {
            const _Float16* bp = woutP + ((size_t)kc * 32 + lane) * 16;
            v16h bW = assemble16(bp, bp + 8);
            racc0 = wmma16(afrag[0][kc], bW, racc0);
            racc1 = wmma16(afrag[1][kc], bW, racc1);
        }
        if (m < 2) {                    // lanes holding valid output columns
            float bo = boutS[m];
#pragma unroll
            for (int r = 0; r < 8; ++r) {
                float v0 = racc0[r] + bo;
                float v1 = racc1[r] + bo;
                out[(size_t)(rowbase +      r + 8*hi) * (PRED_LEN*2) + t*2 + m] = v0;
                out[(size_t)(rowbase + 16 + r + 8*hi) * (PRED_LEN*2) + t*2 + m] = v1;
                relw[m*32 +      r + 8*hi] = v0;
                relw[m*32 + 16 + r + 8*hi] = v1;
            }
        }

        // next x = tanh(rel_pos @ Wsp^T + bsp), packed into A-fragment slots
#pragma unroll
        for (int s = 0; s < 2; ++s) {
            float q0 = relw[     s*16 + m];
            float q1 = relw[32 + s*16 + m];
#pragma unroll
            for (int ee = 0; ee < 8; ++ee) {
                int e = hi * 8 + ee;
                float v = hw_tanh(q0 * wspS[e*2+0] + q1 * wspS[e*2+1] + bspS[e]);
                xfrag[s][ee] = (_Float16)v;   // elems 8..15 (incl. bias 1.0) untouched
            }
        }
    }
}

extern "C" void kernel_launch(void* const* d_in, const int* in_sizes, int n_in,
                              void* d_out, int out_size, void* d_ws, size_t ws_size,
                              hipStream_t stream) {
    (void)in_sizes; (void)n_in; (void)out_size; (void)d_ws; (void)ws_size;
    const float* last_obs_rel = (const float*)d_in[1];
    const float* h0   = (const float*)d_in[2];
    const float* c0   = (const float*)d_in[3];
    const float* Wsp  = (const float*)d_in[4];
    const float* bsp  = (const float*)d_in[5];
    const float* Wih  = (const float*)d_in[6];
    const float* bih  = (const float*)d_in[7];
    const float* Whh  = (const float*)d_in[8];
    const float* bhh  = (const float*)d_in[9];
    const float* Wout = (const float*)d_in[10];
    const float* bout = (const float*)d_in[11];
    float* out = (float*)d_out;

    // >64KB dynamic LDS: raise the attribute limit (gfx1250 WGP has 320KB)
    (void)hipFuncSetAttribute((const void*)lstm_decoder_kernel,
                              hipFuncAttributeMaxDynamicSharedMemorySize,
                              (int)SMEM_BYTES);

    dim3 grid(BATCH / ROWS_PER_BLOCK);   // 256 blocks
    dim3 block(256);                     // 8 waves of 32
    lstm_decoder_kernel<<<grid, block, SMEM_BYTES, stream>>>(
        last_obs_rel, h0, c0, Wsp, bsp, Wih, bih, Whh, bhh, Wout, bout, out);
}
